// SAGEActorCritic_48911087567677
// MI455X (gfx1250) — matrix operand order repaired and checked by reading, not compiled, not run on hardware
//
#include <hip/hip_runtime.h>

#define NN   100000
#define NE   1600000
#define NG   64
#define HIDS 128
#define NACT 32
#define SLOPE 0.01f
#define EPSN  1e-5f

typedef __attribute__((ext_vector_type(16))) __bf16         v16bf;
typedef __attribute__((ext_vector_type(16))) unsigned short v16u;
typedef __attribute__((ext_vector_type(8)))  unsigned short v8u;
typedef __attribute__((ext_vector_type(8)))  float          v8f;

__device__ __forceinline__ unsigned short f2bf(float f) {
  union { float f; unsigned u; } x; x.f = f;
  unsigned r = x.u + 0x7FFFu + ((x.u >> 16) & 1u);   // round-to-nearest-even
  return (unsigned short)(r >> 16);
}
__device__ __forceinline__ float lrelu_f(float v) { return v > 0.f ? v : SLOPE * v; }

// ---------------- one-time setup kernels ----------------

__global__ void k_degrees(const int* __restrict__ src, const int* __restrict__ tgt,
                          float* __restrict__ deg_in, float* __restrict__ deg_out) {
  int e = blockIdx.x * 256 + threadIdx.x;
  if (e >= NE) return;
  atomicAdd(deg_in  + tgt[e], 1.0f);
  atomicAdd(deg_out + src[e], 1.0f);
}

__global__ void k_graph_counts(const int* __restrict__ batch, float* __restrict__ gcount) {
  int v = blockIdx.x * 256 + threadIdx.x;
  if (v >= NN) return;
  atomicAdd(gcount + batch[v], 1.0f);
}

// Convert weights to bf16 AND pre-swizzle into WMMA B-fragment order:
// dest index r (within a layer) = ((nT*4 + kT)*32 + lane)*16 + j
//   lane 0-15  -> N = nT*16 + lane,    K = kT*32 + j        (j = 0..15)
//   lane 16-31 -> N = nT*16 + lane-16, K = kT*32 + 16 + j
// so each lane's 16-half B fragment is 32 contiguous bytes.
__global__ void k_convert_w(const float* __restrict__ wl, const float* __restrict__ wr,
                            unsigned short* __restrict__ wlb, unsigned short* __restrict__ wrb) {
  int i = blockIdx.x * 256 + threadIdx.x;
  if (i >= 5 * HIDS * HIDS) return;
  int L = i / (HIDS * HIDS);
  int r = i % (HIDS * HIDS);
  int j    = r & 15;
  int lane = (r >> 4) & 31;
  int kT   = (r >> 9) & 3;
  int nT   = (r >> 11) & 7;
  int K  = kT * 32 + (lane >> 4) * 16 + j;
  int Nn = nT * 16 + (lane & 15);
  long s = (long)L * HIDS * HIDS + (long)K * HIDS + Nn;
  wlb[i] = f2bf(wl[s]);
  wrb[i] = f2bf(wr[s]);
}

// ---------------- per-layer kernels ----------------

// one wave (32 lanes) per edge; each lane handles 4 contiguous floats
__global__ void k_scatter(const float* __restrict__ h, const int* __restrict__ src,
                          const int* __restrict__ tgt,
                          float* __restrict__ accf, float* __restrict__ accb) {
  int e = blockIdx.x * 8 + (threadIdx.x >> 5);
  if (e >= NE) return;
  int lane = threadIdx.x & 31;
  int s = src[e], t = tgt[e];
  const float4 vs = ((const float4*)(h + (long)s * HIDS))[lane];
  const float4 vt = ((const float4*)(h + (long)t * HIDS))[lane];
  float* af = accf + (long)t * HIDS + lane * 4;
  float* ab = accb + (long)s * HIDS + lane * 4;
  atomicAdd(af + 0, vs.x); atomicAdd(af + 1, vs.y);
  atomicAdd(af + 2, vs.z); atomicAdd(af + 3, vs.w);
  atomicAdd(ab + 0, vt.x); atomicAdd(ab + 1, vt.y);
  atomicAdd(ab + 2, vt.z); atomicAdd(ab + 3, vt.w);
}

// agg = 0.5*(accf/deg_in + accb/deg_out) -> bf16 ; also h -> bf16 (vectorized x4)
__global__ void k_combine(const float* __restrict__ accf, const float* __restrict__ accb,
                          const float* __restrict__ deg_in, const float* __restrict__ deg_out,
                          const float* __restrict__ h,
                          unsigned short* __restrict__ aggb, unsigned short* __restrict__ hb) {
  long i4 = ((long)blockIdx.x * 256 + threadIdx.x) * 4;
  if (i4 >= (long)NN * HIDS) return;
  int v = (int)(i4 / HIDS);
  float di = fmaxf(deg_in[v], 1.0f);
  float dn = fmaxf(deg_out[v], 1.0f);
  float4 f  = *(const float4*)(accf + i4);
  float4 b  = *(const float4*)(accb + i4);
  float4 hh = *(const float4*)(h + i4);
  ushort4 oa, oh;
  oa.x = f2bf(0.5f * (f.x / di + b.x / dn));
  oa.y = f2bf(0.5f * (f.y / di + b.y / dn));
  oa.z = f2bf(0.5f * (f.z / di + b.z / dn));
  oa.w = f2bf(0.5f * (f.w / di + b.w / dn));
  oh.x = f2bf(hh.x); oh.y = f2bf(hh.y); oh.z = f2bf(hh.z); oh.w = f2bf(hh.w);
  *(ushort4*)(aggb + i4) = oa;
  *(ushort4*)(hb   + i4) = oh;
}

// xn = agg@Wl + bl + h@Wr   via v_wmma_f32_16x16x32_bf16
// block = 256 threads = 8 waves; block handles 16 rows, each wave one 16x16 col tile.
// A rows staged into LDS with CDNA5 async global->LDS copies (ASYNCcnt).
__global__ void k_gemm_wmma(const unsigned short* __restrict__ aggb,
                            const unsigned short* __restrict__ hb,
                            const unsigned short* __restrict__ WlSwz,
                            const unsigned short* __restrict__ WrSwz,
                            const float* __restrict__ bl,
                            float* __restrict__ xn) {
  __shared__ __attribute__((aligned(16))) unsigned short sA[16 * HIDS];  // agg rows
  __shared__ __attribute__((aligned(16))) unsigned short sH[16 * HIDS];  // h rows
  const int m0  = blockIdx.x * 16;
  const int tid = threadIdx.x;

  // Async stage: 256 threads x 16B per matrix (4KB each).
  {
    unsigned ldsA = (unsigned)(size_t)&sA[0] + (unsigned)tid * 16u;
    unsigned ldsH = (unsigned)(size_t)&sH[0] + (unsigned)tid * 16u;
    unsigned goff = (unsigned)tid * 16u;                 // byte offset (GVS mode)
    const unsigned short* gA = aggb + (size_t)m0 * HIDS;
    const unsigned short* gH = hb   + (size_t)m0 * HIDS;
    asm volatile("global_load_async_to_lds_b128 %0, %1, %2\n\t"
                 "global_load_async_to_lds_b128 %3, %4, %5"
                 :
                 : "v"(ldsA), "v"(goff), "s"(gA),
                   "v"(ldsH), "v"(goff), "s"(gH)
                 : "memory");
    asm volatile("s_wait_asynccnt 0" ::: "memory");
  }
  __syncthreads();

  const int wave = tid >> 5;
  const int lane = tid & 31;
  const int half = lane >> 4;      // ISA: lanes 0-15 vs 16-31 halves
  const int lm   = lane & 15;

  v8f c = {};
  #pragma unroll
  for (int k0 = 0; k0 < HIDS; k0 += 32) {
    const int klo = k0 + half * 16;       // A elems 0..7  -> K=klo..klo+7
    const int khi = k0 + half * 16 + 8;   // A elems 8..15 -> K=khi..khi+7
    const int kT  = k0 >> 5;
    // B fragment: 32 contiguous bytes per lane from the pre-swizzled weights
    const long wbase = (long)(((wave * 4 + kT) * 32 + lane)) * 16;

    v8u alo = *(const v8u*)(sA + lm * HIDS + klo);
    v8u ahi = *(const v8u*)(sA + lm * HIDS + khi);
    v16u au = __builtin_shufflevector(alo, ahi, 0,1,2,3,4,5,6,7,8,9,10,11,12,13,14,15);
    v8u blo = ((const v8u*)(WlSwz + wbase))[0];
    v8u bhi = ((const v8u*)(WlSwz + wbase))[1];
    v16u bu = __builtin_shufflevector(blo, bhi, 0,1,2,3,4,5,6,7,8,9,10,11,12,13,14,15);
    c = __builtin_amdgcn_wmma_f32_16x16x32_bf16(
            false, __builtin_bit_cast(v16bf, au),
            false, __builtin_bit_cast(v16bf, bu),
            (short)0, c, false, false);

    alo = *(const v8u*)(sH + lm * HIDS + klo);
    ahi = *(const v8u*)(sH + lm * HIDS + khi);
    au  = __builtin_shufflevector(alo, ahi, 0,1,2,3,4,5,6,7,8,9,10,11,12,13,14,15);
    blo = ((const v8u*)(WrSwz + wbase))[0];
    bhi = ((const v8u*)(WrSwz + wbase))[1];
    bu  = __builtin_shufflevector(blo, bhi, 0,1,2,3,4,5,6,7,8,9,10,11,12,13,14,15);
    c = __builtin_amdgcn_wmma_f32_16x16x32_bf16(
            false, __builtin_bit_cast(v16bf, au),
            false, __builtin_bit_cast(v16bf, bu),
            (short)0, c, false, false);
  }

  // D layout: element r -> row m0 + r + 8*half, col n0 + lm
  const int n0 = wave * 16;
  #pragma unroll
  for (int r = 0; r < 8; ++r) {
    int row = m0 + r + 8 * half;
    int col = n0 + lm;
    xn[(long)row * HIDS + col] = c[r] + bl[col];
  }
}

__global__ void k_gsum(const float* __restrict__ xn, const int* __restrict__ batch,
                       float* __restrict__ gsum) {
  long i = (long)blockIdx.x * 256 + threadIdx.x;
  if (i >= (long)NN * HIDS) return;
  int v = (int)(i / HIDS), cidx = (int)(i & (HIDS - 1));
  atomicAdd(gsum + batch[v] * HIDS + cidx, xn[i]);
}

__global__ void k_gvar(const float* __restrict__ xn, const int* __restrict__ batch,
                       const float* __restrict__ gsum, const float* __restrict__ gcount,
                       const float* __restrict__ ms, float* __restrict__ gvar) {
  long i = (long)blockIdx.x * 256 + threadIdx.x;
  if (i >= (long)NN * HIDS) return;
  int v = (int)(i / HIDS), cidx = (int)(i & (HIDS - 1));
  int g = batch[v];
  float cnt  = fmaxf(gcount[g], 1.0f);
  float mean = gsum[g * HIDS + cidx] / cnt;
  float xc   = xn[i] - mean * ms[cidx];
  atomicAdd(gvar + g * HIDS + cidx, xc * xc);
}

__global__ void k_apply(const float* __restrict__ xn, const int* __restrict__ batch,
                        const float* __restrict__ gsum, const float* __restrict__ gvar,
                        const float* __restrict__ gcount,
                        const float* __restrict__ w, const float* __restrict__ b,
                        const float* __restrict__ ms,
                        float* __restrict__ h, int residual) {
  long i = (long)blockIdx.x * 256 + threadIdx.x;
  if (i >= (long)NN * HIDS) return;
  int v = (int)(i / HIDS), cidx = (int)(i & (HIDS - 1));
  int g = batch[v];
  float cnt  = fmaxf(gcount[g], 1.0f);
  float mean = gsum[g * HIDS + cidx] / cnt;
  float var  = gvar[g * HIDS + cidx] / cnt;
  float xc   = xn[i] - mean * ms[cidx];
  float y    = lrelu_f(w[cidx] * xc * rsqrtf(var + EPSN) + b[cidx]);
  h[i] = residual ? (h[i] + y) : y;
}

// ---------------- pooling & heads ----------------

__global__ void k_psum(const float* __restrict__ h, const int* __restrict__ batch,
                       float* __restrict__ esum) {
  long i = (long)blockIdx.x * 256 + threadIdx.x;
  if (i >= (long)NN * HIDS) return;
  int v = (int)(i / HIDS), cidx = (int)(i & (HIDS - 1));
  atomicAdd(esum + batch[v] * HIDS + cidx, h[i]);
}

__global__ void k_heads(const float* __restrict__ esum, const float* __restrict__ gcount,
                        const float* __restrict__ tW, const float* __restrict__ tb,
                        const float* __restrict__ aW1, const float* __restrict__ ab1,
                        const float* __restrict__ aW2, const float* __restrict__ ab2,
                        const float* __restrict__ cW1, const float* __restrict__ cb1,
                        const float* __restrict__ cW2, const float* __restrict__ cb2,
                        float* __restrict__ out) {
  __shared__ float emb[HIDS], t[HIDS], a1[HIDS], c1[HIDS];
  int g = blockIdx.x, c = threadIdx.x;
  float cnt = fmaxf(gcount[g], 1.0f);
  emb[c] = esum[g * HIDS + c] / cnt;
  __syncthreads();
  float acc = tb[c];
  for (int k = 0; k < HIDS; ++k) acc += emb[k] * tW[k * HIDS + c];
  t[c] = lrelu_f(acc);
  __syncthreads();
  float accA = ab1[c], accC = cb1[c];
  for (int k = 0; k < HIDS; ++k) {
    accA += t[k] * aW1[k * HIDS + c];
    accC += t[k] * cW1[k * HIDS + c];
  }
  a1[c] = lrelu_f(accA);
  c1[c] = lrelu_f(accC);
  __syncthreads();
  if (c < NACT) {
    float l = ab2[c];
    for (int k = 0; k < HIDS; ++k) l += a1[k] * aW2[k * NACT + c];
    out[g * NACT + c] = l;                       // logits [NG, NACT]
  }
  if (c == 0) {
    float vv = cb2[0];
    for (int k = 0; k < HIDS; ++k) vv += c1[k] * cW2[k];
    out[NG * NACT + g] = vv;                     // value [NG]
  }
}

// ---------------- launch ----------------

extern "C" void kernel_launch(void* const* d_in, const int* in_sizes, int n_in,
                              void* d_out, int out_size, void* d_ws, size_t ws_size,
                              hipStream_t stream) {
  const float* x        = (const float*)d_in[0];
  const int*   eidx     = (const int*)  d_in[1];
  const int*   batch    = (const int*)  d_in[2];
  const float* lin_l_W  = (const float*)d_in[3];
  const float* lin_l_b  = (const float*)d_in[4];
  const float* lin_r_W  = (const float*)d_in[5];
  const float* norm_w   = (const float*)d_in[6];
  const float* norm_b   = (const float*)d_in[7];
  const float* norm_ms  = (const float*)d_in[8];
  const float* trunk_W  = (const float*)d_in[9];
  const float* trunk_b  = (const float*)d_in[10];
  const float* aW1      = (const float*)d_in[11];
  const float* ab1      = (const float*)d_in[12];
  const float* aW2      = (const float*)d_in[13];
  const float* ab2      = (const float*)d_in[14];
  const float* cW1      = (const float*)d_in[15];
  const float* cb1      = (const float*)d_in[16];
  const float* cW2      = (const float*)d_in[17];
  const float* cb2      = (const float*)d_in[18];
  const int* src = eidx;
  const int* tgt = eidx + NE;
  float* out = (float*)d_out;

  // carve workspace
  char* p = (char*)d_ws;
  auto carve = [&](size_t bytes) -> void* {
    void* r = (void*)p;
    p += (bytes + 255) & ~(size_t)255;
    return r;
  };
  const size_t fbytes = sizeof(float) * (size_t)NN * HIDS;
  float*          h       = (float*)carve(fbytes);
  float*          xn      = (float*)carve(fbytes);
  float*          accf    = (float*)carve(fbytes);
  float*          accb    = (float*)carve(fbytes);
  unsigned short* aggb    = (unsigned short*)carve(sizeof(short) * (size_t)NN * HIDS);
  unsigned short* hb      = (unsigned short*)carve(sizeof(short) * (size_t)NN * HIDS);
  unsigned short* wlb     = (unsigned short*)carve(sizeof(short) * 5 * HIDS * HIDS);
  unsigned short* wrb     = (unsigned short*)carve(sizeof(short) * 5 * HIDS * HIDS);
  float*          deg_in  = (float*)carve(sizeof(float) * NN);
  float*          deg_out = (float*)carve(sizeof(float) * NN);
  float*          gcount  = (float*)carve(sizeof(float) * NG);
  float*          gsum    = (float*)carve(sizeof(float) * NG * HIDS);
  float*          gvar    = (float*)carve(sizeof(float) * NG * HIDS);
  float*          esum    = (float*)carve(sizeof(float) * NG * HIDS);

  const int ELEM_BLKS = (int)(((long)NN * HIDS) / 256);       // 50000 exact
  const int VEC4_BLKS = (int)(((long)NN * HIDS) / 4 / 256);   // 12500 exact

  // one-time setup
  hipMemsetAsync(deg_in,  0, sizeof(float) * NN, stream);
  hipMemsetAsync(deg_out, 0, sizeof(float) * NN, stream);
  hipMemsetAsync(gcount,  0, sizeof(float) * NG, stream);
  k_degrees<<<(NE + 255) / 256, 256, 0, stream>>>(src, tgt, deg_in, deg_out);
  k_graph_counts<<<(NN + 255) / 256, 256, 0, stream>>>(batch, gcount);
  k_convert_w<<<(5 * HIDS * HIDS + 255) / 256, 256, 0, stream>>>(lin_l_W, lin_r_W, wlb, wrb);
  hipMemcpyAsync(h, x, fbytes, hipMemcpyDeviceToDevice, stream);

  for (int L = 0; L < 5; ++L) {
    hipMemsetAsync(accf, 0, fbytes, stream);
    hipMemsetAsync(accb, 0, fbytes, stream);
    k_scatter<<<NE / 8, 256, 0, stream>>>(h, src, tgt, accf, accb);
    k_combine<<<VEC4_BLKS, 256, 0, stream>>>(accf, accb, deg_in, deg_out, h, aggb, hb);
    k_gemm_wmma<<<NN / 16, 256, 0, stream>>>(aggb, hb,
                                             wlb + (size_t)L * HIDS * HIDS,
                                             wrb + (size_t)L * HIDS * HIDS,
                                             lin_l_b + (size_t)L * HIDS, xn);
    hipMemsetAsync(gsum, 0, sizeof(float) * NG * HIDS, stream);
    hipMemsetAsync(gvar, 0, sizeof(float) * NG * HIDS, stream);
    k_gsum<<<ELEM_BLKS, 256, 0, stream>>>(xn, batch, gsum);
    k_gvar<<<ELEM_BLKS, 256, 0, stream>>>(xn, batch, gsum, gcount,
                                          norm_ms + (size_t)L * HIDS, gvar);
    k_apply<<<ELEM_BLKS, 256, 0, stream>>>(xn, batch, gsum, gvar, gcount,
                                           norm_w + (size_t)L * HIDS,
                                           norm_b + (size_t)L * HIDS,
                                           norm_ms + (size_t)L * HIDS,
                                           h, (L >= 2) ? 1 : 0);
  }

  hipMemsetAsync(esum, 0, sizeof(float) * NG * HIDS, stream);
  k_psum<<<ELEM_BLKS, 256, 0, stream>>>(h, batch, esum);
  k_heads<<<NG, HIDS, 0, stream>>>(esum, gcount, trunk_W, trunk_b,
                                   aW1, ab1, aW2, ab2, cW1, cb1, cW2, cb2, out);
}
